// SelfAttention_36670430773696
// MI455X (gfx1250) — matrix operand order repaired and checked by reading, not compiled, not run on hardware
//
#include <hip/hip_runtime.h>

// MI455X (gfx1250) self-attention: bf16 WMMA + TDM (tensor_load_to_lds) staging.
// B=2, T=2048, D=2048, H=16, HD=128.

#define BB 2
#define TT 2048
#define DD 2048
#define HH 16
#define HD 128

typedef __attribute__((ext_vector_type(16))) __bf16 v16bf;
typedef __attribute__((ext_vector_type(8)))  float  v8f;
typedef __attribute__((ext_vector_type(4)))  unsigned int u32x4;
typedef __attribute__((ext_vector_type(4)))  int i32x4;
typedef __attribute__((ext_vector_type(8)))  int i32x8;

struct alignas(8) bf16x4 { __bf16 v[4]; };

__device__ __forceinline__ v8f wmma_bf16(v16bf a, v16bf b, v8f c) {
  // D = A(16x32 bf16) * B(32x16 bf16) + C(16x16 f32)
  return __builtin_amdgcn_wmma_f32_16x16x32_bf16(
      /*neg_a=*/false, a, /*neg_b=*/false, b,
      /*c_mod=*/(short)0, c, /*reuse_a=*/false, /*reuse_b=*/false);
}

// ---------------------------------------------------------------------------
// TDM: issue TENSOR_LOAD_TO_LDS of a 2D tile (tile_w dwords x tile_h rows,
// global row stride stride_dw dwords) into LDS at lds_off, with optional LDS
// row padding (pad_int_code/pad_amt_code per D# group1 encodings; -1 = none).
// D# layout per CDNA5 ISA 8.3/8.4: group0 = {flags, lds_addr, gaddr_lo,
// gaddr_hi|type=2}; group1 packs data_size/pad/tensor dims/tile dims/strides.
// ---------------------------------------------------------------------------
__device__ __forceinline__ void tdm_load_2d(unsigned int lds_off,
                                            const void* gptr,
                                            unsigned int tile_w_dw,
                                            unsigned int tile_h,
                                            unsigned long long stride_dw,
                                            int pad_int_code, int pad_amt_code) {
  unsigned long long ga = (unsigned long long)(uintptr_t)gptr;
  u32x4 g0;
  g0[0] = 1u;                                         // count=1 (valid), user mode
  g0[1] = lds_off;                                    // lds_addr [63:32]
  g0[2] = (unsigned int)(ga & 0xFFFFFFFFull);         // global_addr lo
  g0[3] = (unsigned int)((ga >> 32) & 0x1FFFFFFull)   // global_addr [56:32]
          | (2u << 30);                               // type = 2 ("image")

  unsigned int w0 = (2u << 16);                       // data_size = 2 -> 4B units
  if (pad_amt_code >= 0)
    w0 |= (1u << 20) | ((unsigned int)pad_int_code << 22)
        | ((unsigned int)pad_amt_code << 25);
  i32x8 g1;
  g1[0] = (int)w0;                                    // wg_mask=0, pads
  g1[1] = (int)((tile_w_dw & 0xFFFFu) << 16);         // tensor_dim0[15:0] @ [63:48]
  g1[2] = (int)((tile_w_dw >> 16) | ((tile_h & 0xFFFFu) << 16)); // dim0 hi, dim1 lo
  g1[3] = (int)((tile_h >> 16) | (tile_w_dw << 16));  // dim1 hi, tile_dim0
  g1[4] = (int)tile_h;                                // tile_dim1 (tile_dim2 = 0)
  g1[5] = (int)(stride_dw & 0xFFFFFFFFull);           // tensor_dim0_stride lo32
  g1[6] = (int)((stride_dw >> 32) & 0xFFFFull);       // stride0 hi16 (stride1=0)
  g1[7] = 0;
  i32x4 z4 = {0, 0, 0, 0};
#if __clang_major__ >= 23
  i32x8 z8 = {0, 0, 0, 0, 0, 0, 0, 0};
  __builtin_amdgcn_tensor_load_to_lds(g0, g1, z4, z4, z8, 0);
#else
  __builtin_amdgcn_tensor_load_to_lds(g0, g1, z4, z4, 0);
#endif
}

// ---------------------------------------------------------------------------
// Generic tiled GEMM: C(MxN) = A(MxK) * B(KxN). Block tile 128x128, 8 waves,
// each wave computes a 32x64 region = 2x4 WMMA 16x16 tiles. K step = 32.
// Staging: float4 global loads, bf16x4 (8B) LDS stores.
// ---------------------------------------------------------------------------
template <bool A_F32, bool C_F32>
__global__ __launch_bounds__(256) void gemm_bf16(
    const float* __restrict__ Af, const __bf16* __restrict__ Ab,
    const float* __restrict__ Bw,
    float* __restrict__ Cf, __bf16* __restrict__ Cb,
    int M, int N, int K) {
  __shared__ __bf16 As[128][40];   // 128x32 tile, +8 pad
  __shared__ __bf16 Bs[32][136];   // 32x128 tile, +8 pad

  const int tid  = threadIdx.x;
  const int lane = tid & 31;
  const int wv   = tid >> 5;       // 0..7
  const int m    = lane & 15;
  const int hb   = lane >> 4;      // half of wave
  const int wm   = (wv >> 1) * 32; // wave row offset in 128 tile
  const int wn   = (wv & 1) * 64;  // wave col offset in 128 tile
  const int bm   = blockIdx.y * 128;
  const int bn   = blockIdx.x * 128;

  v8f acc[2][4];
  #pragma unroll
  for (int mt = 0; mt < 2; ++mt)
    #pragma unroll
    for (int nt = 0; nt < 4; ++nt) acc[mt][nt] = {};

  for (int k0 = 0; k0 < K; k0 += 32) {
    // Stage A tile (128x32) -> LDS bf16, 4 elems per thread per pass
    #pragma unroll
    for (int i = 0; i < 4; ++i) {
      int e = i * 256 + tid;            // 4-elem group index
      int r = e >> 3, c = (e & 7) * 4;
      bf16x4 o;
      if (A_F32) {
        float4 v = *(const float4*)&Af[(size_t)(bm + r) * K + k0 + c];
        o.v[0] = (__bf16)v.x; o.v[1] = (__bf16)v.y;
        o.v[2] = (__bf16)v.z; o.v[3] = (__bf16)v.w;
      } else {
        o = *(const bf16x4*)&Ab[(size_t)(bm + r) * K + k0 + c];
      }
      *(bf16x4*)&As[r][c] = o;
    }
    // Stage B tile (32x128) -> LDS bf16
    #pragma unroll
    for (int i = 0; i < 4; ++i) {
      int e = i * 256 + tid;
      int r = e >> 5, c = (e & 31) * 4;
      float4 v = *(const float4*)&Bw[(size_t)(k0 + r) * N + bn + c];
      bf16x4 o;
      o.v[0] = (__bf16)v.x; o.v[1] = (__bf16)v.y;
      o.v[2] = (__bf16)v.z; o.v[3] = (__bf16)v.w;
      *(bf16x4*)&Bs[r][c] = o;
    }
    __syncthreads();

    v16bf afr[2], bfr[4];
    #pragma unroll
    for (int mt = 0; mt < 2; ++mt) {
      int row = wm + mt * 16 + m;
      #pragma unroll
      for (int i = 0; i < 8; ++i) {
        afr[mt][i]     = As[row][8 * hb + i];        // K = i + 8*hb
        afr[mt][8 + i] = As[row][16 + 8 * hb + i];   // K = 16 + 8*hb + i
      }
    }
    #pragma unroll
    for (int nt = 0; nt < 4; ++nt) {
      int col = wn + nt * 16 + m;
      #pragma unroll
      for (int e = 0; e < 16; ++e) bfr[nt][e] = Bs[(e < 8) ? (8 * hb + e) : (16 + 8 * hb + e - 8)][col];
    }
    #pragma unroll
    for (int mt = 0; mt < 2; ++mt)
      #pragma unroll
      for (int nt = 0; nt < 4; ++nt)
        acc[mt][nt] = wmma_bf16(afr[mt], bfr[nt], acc[mt][nt]);
    __syncthreads();
  }

  // Store C: row = r + 8*hb, col = lane&15 within each 16x16 tile
  #pragma unroll
  for (int mt = 0; mt < 2; ++mt)
    #pragma unroll
    for (int nt = 0; nt < 4; ++nt)
      #pragma unroll
      for (int r = 0; r < 8; ++r) {
        int row = bm + wm + mt * 16 + r + 8 * hb;
        int col = bn + wn + nt * 16 + m;
        float v = acc[mt][nt][r];
        if (C_F32) Cf[(size_t)row * N + col] = v;
        else       Cb[(size_t)row * N + col] = (__bf16)v;
      }
}

// ---------------------------------------------------------------------------
// RoPE applied in place on the bf16 qkv buffer (q and k segments).
// qkv layout: [B*T, 3*D]; q cols [0,D), k cols [D,2D).
// ---------------------------------------------------------------------------
__global__ __launch_bounds__(256) void rope_kernel(
    __bf16* __restrict__ qkv, const float* __restrict__ rope) {
  const int PAIRS = BB * TT * HH * (HD / 2);  // 4194304
  int p2 = blockIdx.x * blockDim.x + threadIdx.x;
  if (p2 >= 2 * PAIRS) return;
  int seg = (p2 >= PAIRS) ? 1 : 0;
  int p   = seg ? (p2 - PAIRS) : p2;
  int i   = p & 63;            // pair index within head dim
  int h   = (p >> 6) & 15;
  int bt  = p >> 10;           // b*T + t
  int t   = bt & (TT - 1);
  size_t idx = (size_t)bt * (3 * DD) + seg * DD + h * HD + 2 * i;
  float c = rope[(t * 64 + i) * 2 + 0];
  float s = rope[(t * 64 + i) * 2 + 1];
  float x1 = (float)qkv[idx], x2 = (float)qkv[idx + 1];
  qkv[idx]     = (__bf16)(x1 * c - x2 * s);
  qkv[idx + 1] = (__bf16)(x1 * s + x2 * c);
}

// ---------------------------------------------------------------------------
// Flash attention. Grid (T/128, H, B); block = 256 = 8 waves; each wave owns
// 16 query rows. K/V tiles staged via TDM (tensor_load_to_lds) with the
// +8 bf16 row padding done by the TDM pad feature (interval 64 dw, amount 4 dw).
// ---------------------------------------------------------------------------
__global__ __launch_bounds__(256) void flash_attn(
    const __bf16* __restrict__ qkv, __bf16* __restrict__ out) {
  __shared__ __bf16 Ks[32][136];      // 32 keys x HD, +8 pad (TDM-padded)
  __shared__ __bf16 Vs[32][136];
  __shared__ __bf16 Ps[8][16][32];    // per-wave P tile (16 q x 32 keys)

  const int tid  = threadIdx.x;
  const int lane = tid & 31;
  const int wv   = tid >> 5;
  const int m    = lane & 15;
  const int hb   = lane >> 4;
  const int qt = blockIdx.x, h = blockIdx.y, b = blockIdx.z;
  const size_t rs = 3 * DD;   // qkv row stride (elements)
  const float scale = 0.08838834764831845f;  // 1/sqrt(128)

  // LDS byte offsets for TDM descriptors (low 32 bits of generic LDS pointer).
  const unsigned int ldsK = (unsigned int)(uintptr_t)(void*)Ks;
  const unsigned int ldsV = (unsigned int)(uintptr_t)(void*)Vs;

  // Preload this wave's 16x128 Q as 4 A-fragments (16x32 each).
  const __bf16* qrow =
      qkv + (size_t)(b * TT + qt * 128 + wv * 16 + m) * rs + h * HD;
  v16bf aq[4];
  #pragma unroll
  for (int c4 = 0; c4 < 4; ++c4)
    #pragma unroll
    for (int i = 0; i < 8; ++i) {
      aq[c4][i]     = qrow[32 * c4 + 8 * hb + i];
      aq[c4][8 + i] = qrow[32 * c4 + 16 + 8 * hb + i];
    }

  v8f o[8];
  float mrow[8], lrow[8];
  #pragma unroll
  for (int nt = 0; nt < 8; ++nt) o[nt] = {};
  #pragma unroll
  for (int r = 0; r < 8; ++r) { mrow[r] = -1e30f; lrow[r] = 0.0f; }

  for (int kb = 0; kb < TT / 32; ++kb) {
    __syncthreads();  // previous iteration's LDS reads done
    const int k0 = kb * 32;
    if (wv == 0) {
      // K/V tiles: 32 rows x 128 bf16 (= 64 dwords), row stride 3072 dwords.
      const __bf16* kg = qkv + (size_t)(b * TT + k0) * rs + DD + h * HD;
      const __bf16* vg = qkv + (size_t)(b * TT + k0) * rs + 2 * DD + h * HD;
      tdm_load_2d(ldsK, kg, 64, 32, 3072, /*pad: every 64dw*/ 5, /*+4dw*/ 3);
      tdm_load_2d(ldsV, vg, 64, 32, 3072, 5, 3);
      __builtin_amdgcn_s_wait_tensorcnt(0);
    }
    __syncthreads();

    // S(16x32) = Q(16x128) * K^T(128x32), two 16x16 accumulator tiles.
    v8f sc0 = {}, sc1 = {};
    #pragma unroll
    for (int c4 = 0; c4 < 4; ++c4) {
      v16bf bk0, bk1;
      #pragma unroll
      for (int e = 0; e < 16; ++e) {
        bk0[e] = Ks[m][32 * c4 + ((e < 8) ? (8 * hb + e) : (16 + 8 * hb + e - 8))];       // B col n=key m
        bk1[e] = Ks[16 + m][32 * c4 + ((e < 8) ? (8 * hb + e) : (16 + 8 * hb + e - 8))];  // keys 16..31
      }
      sc0 = wmma_bf16(aq[c4], bk0, sc0);
      sc1 = wmma_bf16(aq[c4], bk1, sc1);
    }

    // Online softmax: row = r + 8*hb; 16 lanes of a half hold one row's cols.
    #pragma unroll
    for (int r = 0; r < 8; ++r) {
      float s0 = sc0[r] * scale, s1 = sc1[r] * scale;
      float mx = fmaxf(s0, s1);
      #pragma unroll
      for (int off = 1; off < 16; off <<= 1)
        mx = fmaxf(mx, __shfl_xor(mx, off, 16));
      float mnew = fmaxf(mrow[r], mx);
      float fac  = __expf(mrow[r] - mnew);
      float p0   = __expf(s0 - mnew);
      float p1   = __expf(s1 - mnew);
      float rsum = p0 + p1;
      #pragma unroll
      for (int off = 1; off < 16; off <<= 1)
        rsum += __shfl_xor(rsum, off, 16);
      lrow[r] = lrow[r] * fac + rsum;
      mrow[r] = mnew;
      #pragma unroll
      for (int nt = 0; nt < 8; ++nt) o[nt][r] *= fac;
      Ps[wv][r + 8 * hb][m]      = (__bf16)p0;  // C-layout -> row-major P
      Ps[wv][r + 8 * hb][16 + m] = (__bf16)p1;
    }
    __syncthreads();

    // O(16x128) += P(16x32) * V(32x128): 8 WMMAs.
    v16bf ap;
    #pragma unroll
    for (int i = 0; i < 8; ++i) {
      ap[i]     = Ps[wv][m][8 * hb + i];
      ap[8 + i] = Ps[wv][m][16 + 8 * hb + i];
    }
    #pragma unroll
    for (int nt = 0; nt < 8; ++nt) {
      v16bf bv;
      #pragma unroll
      for (int e = 0; e < 16; ++e) bv[e] = Vs[(e < 8) ? (8 * hb + e) : (16 + 8 * hb + e - 8)][nt * 16 + m];
      o[nt] = wmma_bf16(ap, bv, o[nt]);
    }
  }

  // Normalize and write bf16 attention output, layout [B*T, D], col = h*HD+d.
  #pragma unroll
  for (int nt = 0; nt < 8; ++nt)
    #pragma unroll
    for (int r = 0; r < 8; ++r) {
      int trow = qt * 128 + wv * 16 + r + 8 * hb;
      float val = o[nt][r] / lrow[r];
      out[(size_t)(b * TT + trow) * DD + h * HD + nt * 16 + m] = (__bf16)val;
    }
}

// ---------------------------------------------------------------------------
extern "C" void kernel_launch(void* const* d_in, const int* in_sizes, int n_in,
                              void* d_out, int out_size, void* d_ws,
                              size_t ws_size, hipStream_t stream) {
  const float* x      = (const float*)d_in[0];  // (B,T,D)
  const float* rope   = (const float*)d_in[1];  // (T,HD/2,2)
  const float* w_attn = (const float*)d_in[2];  // (D,3D)
  const float* w_proj = (const float*)d_in[3];  // (D,D)
  float* out = (float*)d_out;                   // (B,T,D) fp32

  const int M = BB * TT;   // 4096
  __bf16* qkv  = (__bf16*)d_ws;                                   // M x 3D bf16
  __bf16* attn = (__bf16*)((char*)d_ws + (size_t)M * 3 * DD * 2); // M x D bf16

  // 1) QKV = x @ w_attn  (fp32 A -> bf16 LDS; bf16 out)
  gemm_bf16<true, false><<<dim3(3 * DD / 128, M / 128), 256, 0, stream>>>(
      x, nullptr, w_attn, nullptr, qkv, M, 3 * DD, DD);

  // 2) RoPE on q,k segments in place (bf16)
  const int pairs2 = 2 * BB * TT * HH * (HD / 2);
  rope_kernel<<<pairs2 / 256, 256, 0, stream>>>(qkv, rope);

  // 3) Flash attention -> bf16 attn buffer (TDM-staged K/V)
  flash_attn<<<dim3(TT / 128, HH, BB), 256, 0, stream>>>(qkv, attn);

  // 4) out = attn @ w_proj  (bf16 A; fp32 out)
  gemm_bf16<false, true><<<dim3(DD / 128, M / 128), 256, 0, stream>>>(
      nullptr, attn, w_proj, out, nullptr, M, DD, DD);
}
